// MarkovFlow_61993557950526
// MI455X (gfx1250) — compile-verified, matrix operand order checked by reading, not run on hardware
//
#include <hip/hip_runtime.h>

#define T_ 128
#define B_ 512
#define S_ 128
#define D_ 128
#define V_ 50257
#define BC 16            // batches per scan workgroup
#define PADS (S_ + 8)    // padded LDS f16 row stride (breaks 64-bank aliasing)
#define G1 256           // partial-reduction workgroups for column LSE

typedef __attribute__((ext_vector_type(16))) _Float16 v16h;
typedef __attribute__((ext_vector_type(8)))  _Float16 v8h;
typedef __attribute__((ext_vector_type(8)))  float    v8f;

__device__ __forceinline__ v16h cat8(v8h a, v8h b) {
  return __builtin_shufflevector(a, b, 0,1,2,3,4,5,6,7,8,9,10,11,12,13,14,15);
}

// ---------------------------------------------------------------- utilities
__global__ void zero_kernel(float* out) {
  if (threadIdx.x == 0) out[0] = 0.0f;
}

// MhT[s][d] = (transform @ tag_embed^T)[d][s] = dot(transform[d,:], tag_embed[s,:])
__global__ void prep_M(const float* __restrict__ transform,
                       const float* __restrict__ tag_embed,
                       _Float16* __restrict__ MhT) {
  int idx = blockIdx.x * 256 + threadIdx.x;       // 16384 total
  int s = idx >> 7, d = idx & 127;
  float acc = 0.0f;
  #pragma unroll 4
  for (int d2 = 0; d2 < D_; ++d2)
    acc += transform[d * D_ + d2] * tag_embed[s * D_ + d2];
  MhT[s * D_ + d] = (_Float16)acc;
}

// expAhT[j][i] = softmax(tparams[i,:])[j]  (transposed for B-operand loads)
__global__ void prep_expA(const float* __restrict__ tparams,
                          _Float16* __restrict__ expAhT) {
  __shared__ float red[S_];
  int i = blockIdx.x;       // row of tparams
  int j = threadIdx.x;      // 128 threads
  float x = tparams[i * S_ + j];
  red[j] = x; __syncthreads();
  for (int off = 64; off; off >>= 1) {
    if (j < off) red[j] = fmaxf(red[j], red[j + off]);
    __syncthreads();
  }
  float mx = red[0]; __syncthreads();
  float e = __expf(x - mx);
  red[j] = e; __syncthreads();
  for (int off = 64; off; off >>= 1) {
    if (j < off) red[j] += red[j + off];
    __syncthreads();
  }
  expAhT[j * S_ + i] = (_Float16)(e / red[0]);
}

// corr (S x V) -> corrT (V x S), written into the logits buffer
__global__ void transpose_corr(const float* __restrict__ corr,
                               float* __restrict__ corrT) {
  __shared__ float tile[32][33];
  int bx = blockIdx.x;               // V / 32
  int y0 = blockIdx.y * 32;          // S chunk
  int tx = threadIdx.x;              // 0..31
  int x = bx * 32 + tx;              // v for read
  for (int i = threadIdx.y; i < 32; i += 8)
    tile[i][tx] = (x < V_) ? corr[(long)(y0 + i) * V_ + x] : 0.0f;
  __syncthreads();
  for (int i = threadIdx.y; i < 32; i += 8) {
    int vv = bx * 32 + i;
    if (vv < V_) corrT[(long)vv * S_ + y0 + tx] = tile[tx][i];
  }
}

// logits[v][s] = word_embed[v,:] @ M[:,s] + corrT[v][s]   (corrT pre-staged in logits)
__global__ void logits_kernel(const float* __restrict__ word_embed,
                              const _Float16* __restrict__ MhT,
                              float* __restrict__ logits) {
  const int tid  = threadIdx.x;
  const int lane = tid & 31;
  const int wave = tid >> 5;             // 0..7 -> S tile
  const int v0   = blockIdx.x * 16;
  const int s0   = wave * 16;
  const int m_   = lane & 15;
  const int kb   = (lane >> 4) * 8;
  int row = v0 + m_; if (row >= V_) row = V_ - 1;       // clamp; store masked
  const float*    arow = word_embed + (long)row * D_;
  const _Float16* brow = MhT + (s0 + m_) * D_;

  v8f c = {};
  #pragma unroll
  for (int kk = 0; kk < D_; kk += 32) {
    const float4* a4 = (const float4*)(arow + kk + kb);
    float4 f0 = a4[0], f1 = a4[1];               // K = kk+kb .. +7
    const float4* a4h = (const float4*)(arow + kk + kb + 16);
    float4 g0 = a4h[0], g1 = a4h[1];             // K = kk+kb+16 .. +23
    v8h alo, ahi;
    alo[0]=(_Float16)f0.x; alo[1]=(_Float16)f0.y; alo[2]=(_Float16)f0.z; alo[3]=(_Float16)f0.w;
    alo[4]=(_Float16)f1.x; alo[5]=(_Float16)f1.y; alo[6]=(_Float16)f1.z; alo[7]=(_Float16)f1.w;
    ahi[0]=(_Float16)g0.x; ahi[1]=(_Float16)g0.y; ahi[2]=(_Float16)g0.z; ahi[3]=(_Float16)g0.w;
    ahi[4]=(_Float16)g1.x; ahi[5]=(_Float16)g1.y; ahi[6]=(_Float16)g1.z; ahi[7]=(_Float16)g1.w;
    v8h blo = *(const v8h*)(brow + kk + kb);
    v8h bhi = *(const v8h*)(brow + kk + kb + 16);
    c = __builtin_amdgcn_wmma_f32_16x16x32_f16(false, cat8(alo, ahi),
                                               false, cat8(blo, bhi),
                                               (short)0, c, false, false);
  }
  #pragma unroll
  for (int r = 0; r < 8; ++r) {
    int M = v0 + r + ((lane >> 4) << 3);
    if (M < V_) {
      long o = (long)M * S_ + s0 + m_;
      logits[o] = c[r] + logits[o];              // add pre-staged correction
    }
  }
}

// online per-column (over V) max/sumexp partials; 128 threads = 1 column each
__global__ void col_lse_partial(const float* __restrict__ logits,
                                float* __restrict__ partialM,
                                float* __restrict__ partialS) {
  int j = threadIdx.x;
  int wg = blockIdx.x;
  long rows_per = (V_ + G1 - 1) / G1;
  long v0 = (long)wg * rows_per;
  long v1 = v0 + rows_per; if (v1 > V_) v1 = V_;
  float m = -INFINITY, s = 0.0f;
  for (long v = v0; v < v1; ++v) {
    float x = logits[v * S_ + j];
    if (x > m) { s = s * __expf(m - x) + 1.0f; m = x; }
    else       { s += __expf(x - m); }
  }
  partialM[wg * S_ + j] = m;
  partialS[wg * S_ + j] = s;
}

__global__ void col_lse_final(const float* __restrict__ partialM,
                              const float* __restrict__ partialS,
                              float* __restrict__ colNorm) {
  int j = threadIdx.x;
  float M = -INFINITY;
  for (int w = 0; w < G1; ++w) M = fmaxf(M, partialM[w * S_ + j]);
  float s = 0.0f;
  for (int w = 0; w < G1; ++w) s += partialS[w * S_ + j] * __expf(partialM[w * S_ + j] - M);
  colNorm[j] = M + __logf(s);
}

// -------------------------------------------------- HMM forward scan (WMMA)
__global__ void scan_kernel(const int* __restrict__ words,
                            const float* __restrict__ masks,
                            const float* __restrict__ logits,
                            const float* __restrict__ colNorm,
                            const _Float16* __restrict__ expAhT,
                            float* out) {
  __shared__ __attribute__((aligned(16))) float    alpha[BC][S_];   // 8 KB
  __shared__ __attribute__((aligned(16))) _Float16 Ph[BC][PADS];    // ~4.3 KB
  __shared__ __attribute__((aligned(16))) _Float16 AT[S_][PADS];    // ~34 KB
  __shared__ float rowmax[BC];
  __shared__ float red[BC][16];

  const int tid  = threadIdx.x;      // 256 threads = 8 waves
  const int lane = tid & 31;
  const int wave = tid >> 5;
  const int b0   = blockIdx.x * BC;  // this WG's batch chunk
  const int n0   = wave * 16;        // this wave's state-column tile
  const int m_   = lane & 15;
  const int kb   = (lane >> 4) * 8;
  const float LOGPI = -4.852030263919617f;   // log(1/128)

  for (int idx = tid; idx < S_ * S_; idx += 256) {
    int n = idx >> 7, k = idx & 127;
    AT[n][k] = expAhT[idx];
  }
  for (int idx = tid; idx < BC * S_; idx += 256) {
    int r = idx >> 7, j = idx & 127;
    int w = words[0 * B_ + b0 + r];
    alpha[r][j] = LOGPI + logits[(long)w * S_ + j] - colNorm[j];
  }
  __syncthreads();

  // loop-invariant per-lane values
  const int   N    = n0 + m_;                 // this lane's state column
  const float cn   = colNorm[N];
  const int   Mofs = (lane >> 4) << 3;        // 0 or 8

  for (int t = 1; t < T_; ++t) {
    // ---- prefetch em + mask for this step into registers (independent of alpha):
    //      overlaps the rowmax / exp / WMMA phases below
    float emv[8], mskv[8];
    #pragma unroll
    for (int r8 = 0; r8 < 8; ++r8) {
      int b = b0 + r8 + Mofs;
      int w = words[t * B_ + b];
      emv[r8]  = logits[(long)w * S_ + N];
      mskv[r8] = masks[t * B_ + b];
    }

    // per-batch row max of alpha
    {
      int r = tid >> 4, cidx = tid & 15;
      float mx = -INFINITY;
      for (int j = cidx; j < S_; j += 16) mx = fmaxf(mx, alpha[r][j]);
      red[r][cidx] = mx;
    }
    __syncthreads();
    if (tid < BC) {
      float mx = red[tid][0];
      for (int cidx = 1; cidx < 16; ++cidx) mx = fmaxf(mx, red[tid][cidx]);
      rowmax[tid] = mx;
    }
    __syncthreads();
    // P = exp(alpha - rowmax) in f16
    for (int idx = tid; idx < BC * S_; idx += 256) {
      int r = idx >> 7, j = idx & 127;
      Ph[r][j] = (_Float16)__expf(alpha[r][j] - rowmax[r]);
    }
    __syncthreads();

    // 16x16 tile: (P @ expA) over K=128 in 4 WMMA steps
    v8f c = {};
    #pragma unroll
    for (int kk = 0; kk < S_; kk += 32) {
      v8h alo = *(const v8h*)&Ph[m_][kk + kb];
      v8h ahi = *(const v8h*)&Ph[m_][kk + kb + 16];
      v8h blo = *(const v8h*)&AT[n0 + m_][kk + kb];
      v8h bhi = *(const v8h*)&AT[n0 + m_][kk + kb + 16];
      c = __builtin_amdgcn_wmma_f32_16x16x32_f16(false, cat8(alo, ahi),
                                                 false, cat8(blo, bhi),
                                                 (short)0, c, false, false);
    }

    // alpha[b,j] = mask * (rowmax + log(c) + em - colNorm) + (1-mask) * alpha
    #pragma unroll
    for (int r8 = 0; r8 < 8; ++r8) {
      int M = r8 + Mofs;
      float nv = rowmax[M] + __logf(c[r8]) + emv[r8] - cn;
      alpha[M][N] = mskv[r8] * nv + (1.0f - mskv[r8]) * alpha[M][N];
    }
    __syncthreads();
  }

  // objective: sum_b logsumexp_j alpha[b,j]
  {
    int r = tid >> 4, cidx = tid & 15;
    float mx = -INFINITY;
    for (int j = cidx; j < S_; j += 16) mx = fmaxf(mx, alpha[r][j]);
    red[r][cidx] = mx;
  }
  __syncthreads();
  if (tid < BC) {
    float mx = red[tid][0];
    for (int cidx = 1; cidx < 16; ++cidx) mx = fmaxf(mx, red[tid][cidx]);
    rowmax[tid] = mx;
  }
  __syncthreads();
  {
    int r = tid >> 4, cidx = tid & 15;
    float s = 0.0f;
    for (int j = cidx; j < S_; j += 16) s += __expf(alpha[r][j] - rowmax[r]);
    red[r][cidx] = s;
  }
  __syncthreads();
  if (tid == 0) {
    float tot = 0.0f;
    for (int r = 0; r < BC; ++r) {
      float s = 0.0f;
      for (int cidx = 0; cidx < 16; ++cidx) s += red[r][cidx];
      tot += rowmax[r] + __logf(s);
    }
    atomicAdd(out, -tot);       // objective is negated
  }
}

// ------------------------------------------------------------------ launch
extern "C" void kernel_launch(void* const* d_in, const int* in_sizes, int n_in,
                              void* d_out, int out_size, void* d_ws, size_t ws_size,
                              hipStream_t stream) {
  (void)in_sizes; (void)n_in; (void)out_size; (void)ws_size;
  const int*   words      = (const int*)d_in[0];
  const float* masks      = (const float*)d_in[1];
  const float* tparams    = (const float*)d_in[2];
  const float* tag_embed  = (const float*)d_in[3];
  const float* word_embed = (const float*)d_in[4];
  const float* transform  = (const float*)d_in[5];
  const float* corr       = (const float*)d_in[6];
  float* out = (float*)d_out;

  char* ws = (char*)d_ws;
  float* logits    = (float*)ws;     ws += (size_t)V_ * S_ * sizeof(float);   // 25.7 MB
  _Float16* MhT    = (_Float16*)ws;  ws += (size_t)S_ * D_ * sizeof(_Float16);
  _Float16* expAhT = (_Float16*)ws;  ws += (size_t)S_ * S_ * sizeof(_Float16);
  float* partialM  = (float*)ws;     ws += (size_t)G1 * S_ * sizeof(float);
  float* partialS  = (float*)ws;     ws += (size_t)G1 * S_ * sizeof(float);
  float* colNorm   = (float*)ws;

  zero_kernel<<<1, 32, 0, stream>>>(out);
  prep_M<<<(S_ * D_) / 256, 256, 0, stream>>>(transform, tag_embed, MhT);
  prep_expA<<<S_, S_, 0, stream>>>(tparams, expAhT);
  transpose_corr<<<dim3((V_ + 31) / 32, S_ / 32), dim3(32, 8), 0, stream>>>(corr, logits);
  logits_kernel<<<(V_ + 15) / 16, 256, 0, stream>>>(word_embed, MhT, logits);
  col_lse_partial<<<G1, S_, 0, stream>>>(logits, partialM, partialS);
  col_lse_final<<<1, S_, 0, stream>>>(partialM, partialS, colNorm);
  scan_kernel<<<B_ / BC, 256, 0, stream>>>(words, masks, logits, colNorm, expAhT, out);
}
